// BERT_BiGRU_CRF_11269994185173
// MI455X (gfx1250) — matrix operand order repaired and checked by reading, not compiled, not run on hardware
//
#include <hip/hip_runtime.h>
#include <hip/hip_bf16.h>
#include <math.h>

#define B_   128
#define T_   512
#define HID_ 768
#define G_   192   // 3*H
#define H_   64
#define NL   9

typedef __attribute__((ext_vector_type(16))) __bf16 v16bf;
typedef __attribute__((ext_vector_type(8)))  float  v8f;
typedef __attribute__((ext_vector_type(2)))  float  v2f;

union BF16x16 { v16bf v; uint4 q[2]; unsigned int u[8]; };

__device__ __forceinline__ unsigned short f2bf(float x) {
  unsigned int u = __float_as_uint(x);
  u += 0x7fffu + ((u >> 16) & 1u);   // round-to-nearest-even
  return (unsigned short)(u >> 16);
}

__device__ __forceinline__ v8f wmma_bf16(v16bf a, v16bf b, v8f c) {
  // D = A(16x32 bf16) * B(32x16 bf16) + C(16x16 f32)
  return __builtin_amdgcn_wmma_f32_16x16x32_bf16(false, a, false, b, (short)0, c, false, false);
}

// ---------------------------------------------------------------- K0: fp32 -> bf16
__global__ void k_cvt_bf16(const float* __restrict__ in, unsigned short* __restrict__ out,
                           long long n) {
  long long i = ((long long)blockIdx.x * blockDim.x + threadIdx.x) * 8;
  if (i >= n) return;
  if (i + 8 <= n) {
    const float4 a = *(const float4*)(in + i);
    const float4 b = *(const float4*)(in + i + 4);
    uint4 o;
    o.x = (unsigned)f2bf(a.x) | ((unsigned)f2bf(a.y) << 16);
    o.y = (unsigned)f2bf(a.z) | ((unsigned)f2bf(a.w) << 16);
    o.z = (unsigned)f2bf(b.x) | ((unsigned)f2bf(b.y) << 16);
    o.w = (unsigned)f2bf(b.z) | ((unsigned)f2bf(b.w) << 16);
    *(uint4*)(out + i) = o;
  } else {
    for (long long k = i; k < n; k++) out[k] = f2bf(in[k]);
  }
}

// ---------------------------------------------------------------- K1: xp = X @ W_ih^T + b_ih
// grid 4096 (16-row tiles of the 65536 (b,t) rows), 256 threads = 8 waves.
// Wave w owns (dir,n)-tiles 3w..3w+2 of 24 total (12 per direction).
__global__ __launch_bounds__(256)
void k1_proj(const unsigned short* __restrict__ Xbf,       // [B*T][768] bf16
             const unsigned short* __restrict__ Wihbf,     // [2][192][768] bf16
             const float* __restrict__ b_ih_f, const float* __restrict__ b_ih_b,
             float* __restrict__ xp)                       // [2][T][B][192] f32
{
  const int lane = threadIdx.x & 31;
  const int w    = threadIdx.x >> 5;
  const int rowbase = blockIdx.x * 16;
  const int m    = lane & 15;
  const int klo  = (lane >> 4) * 16;   // A half-K offset
  const int kloB = (lane >> 4) * 16;   // B half-K offset

  const unsigned short* aRow = Xbf + (size_t)(rowbase + m) * HID_ + klo;

  const unsigned short* bRow[3];
  int nIdx[3], dirIdx[3];
  #pragma unroll
  for (int i = 0; i < 3; i++) {
    int nt  = w * 3 + i;
    int dir = nt / 12;
    int n   = (nt % 12) * 16 + m;
    dirIdx[i] = dir; nIdx[i] = n;
    bRow[i] = Wihbf + ((size_t)dir * G_ + n) * HID_ + kloB;
  }

  v8f acc[3];
  #pragma unroll
  for (int i = 0; i < 3; i++)
    #pragma unroll
    for (int v = 0; v < 8; v++) acc[i][v] = 0.f;

  for (int kk = 0; kk < HID_; kk += 32) {
    // stream-ahead hint on the HBM-dominant A stream (global_prefetch_b8;
    // speculative, silently dropped past the buffer end)
    __builtin_prefetch(aRow + kk + 256, 0, 1);
    BF16x16 A;
    A.q[0] = *(const uint4*)(aRow + kk);
    A.q[1] = *(const uint4*)(aRow + kk + 8);
    #pragma unroll
    for (int i = 0; i < 3; i++) {
      BF16x16 Bf;
      Bf.q[0] = *(const uint4*)(bRow[i] + kk);
      Bf.q[1] = *(const uint4*)(bRow[i] + kk + 8);
      acc[i] = wmma_bf16(A.v, Bf.v, acc[i]);
    }
  }

  #pragma unroll
  for (int i = 0; i < 3; i++) {
    const float* bias = dirIdx[i] ? b_ih_b : b_ih_f;
    const float  bv   = bias[nIdx[i]];
    float* outd = xp + (size_t)dirIdx[i] * T_ * B_ * G_;
    #pragma unroll
    for (int v = 0; v < 8; v++) {
      int row = v + (lane >> 4) * 8;
      int r   = rowbase + row;
      int bb  = r >> 9;      // / T_
      int tt  = r & (T_ - 1);
      outd[((size_t)tt * B_ + bb) * G_ + nIdx[i]] = acc[i][v] + bv;
    }
  }
}

// ---------------------------------------------------------------- K2: BiGRU recurrence
// grid 16 = {dir 0/1} x {8 batch-groups of 16}, 128 threads = 4 waves.
__global__ __launch_bounds__(128)
void k2_gru(const float* __restrict__ xp,                  // [2][T][B][192]
            const unsigned short* __restrict__ Whhbf,      // [2][192][64] bf16
            const float* __restrict__ b_hh_f, const float* __restrict__ b_hh_b,
            const int* __restrict__ length,
            float* __restrict__ feat)                      // [B][T][128]
{
  __shared__ float          hg_s[16 * G_];
  __shared__ float          h_s[16 * H_];
  __shared__ unsigned short hbf_s[16 * H_];
  __shared__ float          bhh_s[G_];

  const int dir  = blockIdx.x & 1;
  const int bg   = blockIdx.x >> 1;
  const int tid  = threadIdx.x;
  const int lane = tid & 31;
  const int w    = tid >> 5;

  for (int i = tid; i < 16 * H_; i += 128) { h_s[i] = 0.f; hbf_s[i] = 0; }
  {
    const float* bh = dir ? b_hh_b : b_hh_f;
    for (int i = tid; i < G_; i += 128) bhh_s[i] = bh[i];
  }

  const int m    = lane & 15;
  const int klo  = (lane >> 4) * 16;
  const int kloB = (lane >> 4) * 16;

  // W_hh bf16 fragments resident in registers for all 512 steps
  BF16x16 bf[3][2];
  #pragma unroll
  for (int i = 0; i < 3; i++) {
    int n = (w * 3 + i) * 16 + m;
    const unsigned short* rowp = Whhbf + ((size_t)dir * G_ + n) * H_;
    #pragma unroll
    for (int s = 0; s < 2; s++) {
      bf[i][s].q[0] = *(const uint4*)(rowp + s * 32 + kloB);
      bf[i][s].q[1] = *(const uint4*)(rowp + s * 32 + kloB + 8);
    }
  }
  __syncthreads();

  const int   mg    = tid >> 3;          // gate-phase row 0..15
  const int   j0    = (tid & 7) * 8;     // gate-phase col base
  const int   bglob = bg * 16 + mg;
  const int   len   = length[bglob];
  const float* xpd0 = xp + (size_t)dir * T_ * B_ * G_ + (size_t)bglob * G_;

  for (int step = 0; step < T_; step++) {
    const int t = dir ? (T_ - 1 - step) : step;

    // --- hg = h @ W_hh^T  (WMMA phase) ---
    BF16x16 afr[2];
    #pragma unroll
    for (int s = 0; s < 2; s++) {
      afr[s].q[0] = *(const uint4*)&hbf_s[m * H_ + s * 32 + klo];
      afr[s].q[1] = *(const uint4*)&hbf_s[m * H_ + s * 32 + 8 + klo];
    }
    #pragma unroll
    for (int i = 0; i < 3; i++) {
      v8f acc;
      #pragma unroll
      for (int v = 0; v < 8; v++) acc[v] = 0.f;
      acc = wmma_bf16(afr[0].v, bf[i][0].v, acc);
      acc = wmma_bf16(afr[1].v, bf[i][1].v, acc);
      const int n = (w * 3 + i) * 16 + m;
      #pragma unroll
      for (int v = 0; v < 8; v++) {
        int row = v + (lane >> 4) * 8;
        hg_s[row * G_ + n] = acc[v];
      }
    }
    __syncthreads();

    // --- gates ---
    const float* xpt  = xpd0 + (size_t)t * B_ * G_;
    // prefetch next step's xp row (recurrence critical path)
    {
      const int tn = dir ? (t - 1) : (t + 1);
      if (tn >= 0 && tn < T_) __builtin_prefetch(xpd0 + (size_t)tn * B_ * G_ + j0, 0, 1);
    }
    const float  msk  = (t < len) ? 1.0f : 0.0f;
    float* fout = feat + ((size_t)bglob * T_ + t) * 128 + dir * H_;
    #pragma unroll
    for (int q = 0; q < 8; q++) {
      const int j = j0 + q;
      const float xr = xpt[j], xz = xpt[H_ + j], xn = xpt[2 * H_ + j];
      const float hr = hg_s[mg * G_ + j]          + bhh_s[j];
      const float hz = hg_s[mg * G_ + H_ + j]     + bhh_s[H_ + j];
      const float hn = hg_s[mg * G_ + 2 * H_ + j] + bhh_s[2 * H_ + j];
      const float r  = 1.f / (1.f + expf(-(xr + hr)));
      const float z  = 1.f / (1.f + expf(-(xz + hz)));
      const float nn = tanhf(xn + r * hn);
      const float hold = h_s[mg * H_ + j];
      const float hnew = (1.f - z) * nn + z * hold;
      const float h    = (t < len) ? hnew : hold;
      h_s[mg * H_ + j]   = h;
      hbf_s[mg * H_ + j] = f2bf(h);
      fout[j] = h * msk;
    }
    __syncthreads();
  }
}

// ---------------------------------------------------------------- K3: em = feat @ W_lin^T + b_lin
// grid 512 x 256 threads (8 waves); wave = one 16-row tile; fp32 WMMA 16x16x4
// (UNCONDITIONAL now: compile success == definitive proof the f32 WMMA lowers).
__global__ __launch_bounds__(256)
void k3_em(const float* __restrict__ feat, const float* __restrict__ Wlin,
           const float* __restrict__ blin, float* __restrict__ em)
{
  const int lane    = threadIdx.x & 31;
  const int w       = threadIdx.x >> 5;
  const int rowbase = (blockIdx.x * 8 + w) * 16;
  const int m       = lane & 15;
  const int kq0     = (lane >> 4) * 2;

  v8f acc;
  #pragma unroll
  for (int v = 0; v < 8; v++) acc[v] = 0.f;

  const float* arow  = feat + (size_t)(rowbase + m) * 128;
  const int    nrow  = (m < NL) ? m : (NL - 1);   // clamp, keep EXEC uniform
  const float  scale = (m < NL) ? 1.f : 0.f;
  const float* brow  = Wlin + (size_t)nrow * 128;

  for (int kk = 0; kk < 128; kk += 4) {
    v2f a; a[0] = arow[kk + kq0];         a[1] = arow[kk + kq0 + 1];
    v2f b; b[0] = brow[kk + kq0] * scale; b[1] = brow[kk + kq0 + 1] * scale;
    acc = __builtin_amdgcn_wmma_f32_16x16x4_f32(false, a, false, b, (short)0, acc, false, false);
  }
  if (m < NL) {
    const float bv = blin[m];
    #pragma unroll
    for (int v = 0; v < 8; v++) {
      int row = v + (lane >> 4) * 8;
      em[(size_t)(rowbase + row) * NL + m] = acc[v] + bv;
    }
  }
}

// ---------------------------------------------------------------- K4: CRF (one wave per batch)
__global__ __launch_bounds__(32)
void k4_crf(const float* __restrict__ em, const int* __restrict__ label,
            const int* __restrict__ length,
            const float* __restrict__ start_trans, const float* __restrict__ end_trans,
            const float* __restrict__ trans, float* __restrict__ sb)
{
  const int b   = blockIdx.x;
  const int j   = threadIdx.x;
  const int len = length[b];
  const float* emb = em + (size_t)b * T_ * NL;
  const int*   lab = label + (size_t)b * T_;
  const bool act = (j < NL);

  float trj[NL];
  #pragma unroll
  for (int i = 0; i < NL; i++) trj[i] = act ? trans[i * NL + j] : 0.f;

  float alpha = act ? (start_trans[j] + emb[j]) : -1e30f;

  for (int t = 1; t < len; t++) {
    float ai[NL];
    #pragma unroll
    for (int i = 0; i < NL; i++) ai[i] = __shfl(alpha, i, 32);
    float mx = -1e30f;
    #pragma unroll
    for (int i = 0; i < NL; i++) mx = fmaxf(mx, ai[i] + trj[i]);
    float s = 0.f;
    #pragma unroll
    for (int i = 0; i < NL; i++) s += expf(ai[i] + trj[i] - mx);
    const float ev  = act ? emb[(size_t)t * NL + j] : 0.f;
    const float nxt = ev + mx + logf(s);
    alpha = act ? nxt : alpha;
  }

  // logZ = logsumexp_j(alpha_j + end_trans_j)
  float x = act ? (alpha + end_trans[j]) : -1e30f;
  float mx = x;
  #pragma unroll
  for (int off = 16; off > 0; off >>= 1) mx = fmaxf(mx, __shfl_xor(mx, off, 32));
  float s = act ? expf(x - mx) : 0.f;
  #pragma unroll
  for (int off = 16; off > 0; off >>= 1) s += __shfl_xor(s, off, 32);
  const float logZ = mx + logf(s);

  // gold-path score: lane-strided partial sums over t in [1, len)
  float part = 0.f;
  for (int t = 1 + j; t < len; t += 32) {
    const int lp = lab[t - 1], lc = lab[t];
    part += trans[lp * NL + lc] + emb[(size_t)t * NL + lc];
  }
  #pragma unroll
  for (int off = 16; off > 0; off >>= 1) part += __shfl_xor(part, off, 32);

  if (j == 0) {
    const int l0 = lab[0];
    float score = part + start_trans[l0] + emb[l0] + end_trans[lab[len - 1]];
    sb[b] = score - logZ;
  }
}

// ---------------------------------------------------------------- K5: loss = -mean
__global__ __launch_bounds__(128)
void k5_red(const float* __restrict__ sb, float* __restrict__ out)
{
  __shared__ float sh[128];
  const int t = threadIdx.x;
  sh[t] = sb[t];
  __syncthreads();
  for (int s = 64; s > 0; s >>= 1) {
    if (t < s) sh[t] += sh[t + s];
    __syncthreads();
  }
  if (t == 0) out[0] = -(sh[0] / (float)B_);
}

// ================================================================ host
extern "C" void kernel_launch(void* const* d_in, const int* in_sizes, int n_in,
                              void* d_out, int out_size, void* d_ws, size_t ws_size,
                              hipStream_t stream)
{
  const int*   length  = (const int*)  d_in[0];
  const float* w2v     = (const float*)d_in[1];
  // d_in[2] = mask (bool) -- recomputed from length on device
  const int*   label   = (const int*)  d_in[3];
  const float* W_ih_f  = (const float*)d_in[4];
  const float* W_hh_f  = (const float*)d_in[5];
  const float* b_ih_f  = (const float*)d_in[6];
  const float* b_hh_f  = (const float*)d_in[7];
  const float* W_ih_b  = (const float*)d_in[8];
  const float* W_hh_b  = (const float*)d_in[9];
  const float* b_ih_b  = (const float*)d_in[10];
  const float* b_hh_b  = (const float*)d_in[11];
  const float* W_lin   = (const float*)d_in[12];
  const float* b_lin   = (const float*)d_in[13];
  const float* start_t = (const float*)d_in[14];
  const float* end_t   = (const float*)d_in[15];
  const float* trans   = (const float*)d_in[16];

  char* ws = (char*)d_ws;
  float* xp   = (float*)ws;  ws += (size_t)2 * T_ * B_ * G_ * sizeof(float);   // 100.7 MB
  float* feat = (float*)ws;  ws += (size_t)B_ * T_ * 128 * sizeof(float);      //  33.6 MB
  float* em   = (float*)ws;  ws += (size_t)B_ * T_ * NL * sizeof(float);       //   2.4 MB
  float* sb   = (float*)ws;  ws += 512;
  unsigned short* Xbf   = (unsigned short*)ws; ws += (size_t)B_ * T_ * HID_ * 2; // 100.7 MB
  unsigned short* Wihbf = (unsigned short*)ws; ws += (size_t)2 * G_ * HID_ * 2;
  unsigned short* Whhbf = (unsigned short*)ws; ws += (size_t)2 * G_ * H_ * 2;

  // K0: fp32 -> bf16 conversions (once per call)
  const long long nX = (long long)B_ * T_ * HID_;
  k_cvt_bf16<<<(unsigned)((nX / 8 + 255) / 256), 256, 0, stream>>>(w2v, Xbf, nX);
  const long long nWih = (long long)G_ * HID_;
  const long long nWhh = (long long)G_ * H_;
  k_cvt_bf16<<<(unsigned)((nWih / 8 + 255) / 256), 256, 0, stream>>>(W_ih_f, Wihbf, nWih);
  k_cvt_bf16<<<(unsigned)((nWih / 8 + 255) / 256), 256, 0, stream>>>(W_ih_b, Wihbf + nWih, nWih);
  k_cvt_bf16<<<(unsigned)((nWhh / 8 + 255) / 256), 256, 0, stream>>>(W_hh_f, Whhbf, nWhh);
  k_cvt_bf16<<<(unsigned)((nWhh / 8 + 255) / 256), 256, 0, stream>>>(W_hh_b, Whhbf + nWhh, nWhh);

  // K1: input projections (bf16 WMMA, memory-bound)
  k1_proj<<<(B_ * T_) / 16, 256, 0, stream>>>(Xbf, Wihbf, b_ih_f, b_ih_b, xp);

  // K2: bidirectional GRU recurrence (WMMA per step, weights in registers)
  k2_gru<<<16, 128, 0, stream>>>(xp, Whhbf, b_hh_f, b_hh_b, length, feat);

  // K3: emissions (fp32 WMMA 16x16x4)
  k3_em<<<(B_ * T_) / (16 * 8), 256, 0, stream>>>(feat, W_lin, b_lin, em);

  // K4: CRF forward + gold score per batch
  k4_crf<<<B_, 32, 0, stream>>>(em, label, length, start_t, end_t, trans, sb);

  // K5: final reduction
  k5_red<<<1, 128, 0, stream>>>(sb, (float*)d_out);
}
